// VisionMamba_43387759624399
// MI455X (gfx1250) — compile-verified, matrix operand order checked by reading, not run on hardware
//
#include <hip/hip_runtime.h>
#include <hip/hip_bf16.h>
#include <math.h>

typedef __attribute__((ext_vector_type(16))) _Float16 v16h;
typedef __attribute__((ext_vector_type(8)))  float    v8f;

#define D_MODEL 192
#define DEPTH   4
#define D_INNER 384
#define D_STATE 16
#define D_CONV  4
#define DT_RANK 12
#define BATCH   16
#define SEQLEN  2048
#define NTOK    (BATCH * SEQLEN)
#define XDBL_LD 48   // 44 padded to 48

// ---------------------------------------------------------------------------
// Load one wave32 WMMA f16 fragment (A-layout; B mirrors with lane=column).
// p points at row_base + k0 + kbase; element e<8 -> p[e], e>=8 -> p[e+8].
// Guard-free: 4x global_load_b128 + packed f32->f16 converts.
// ---------------------------------------------------------------------------
__device__ __forceinline__ v16h load_frag(const float* __restrict__ p) {
    const float4* q  = (const float4*)p;
    const float4* q2 = (const float4*)(p + 16);
    float4 x0 = q[0], x1 = q[1], x2 = q2[0], x3 = q2[1];
    v16h r;
    r[0]  = (_Float16)x0.x; r[1]  = (_Float16)x0.y;
    r[2]  = (_Float16)x0.z; r[3]  = (_Float16)x0.w;
    r[4]  = (_Float16)x1.x; r[5]  = (_Float16)x1.y;
    r[6]  = (_Float16)x1.z; r[7]  = (_Float16)x1.w;
    r[8]  = (_Float16)x2.x; r[9]  = (_Float16)x2.y;
    r[10] = (_Float16)x2.z; r[11] = (_Float16)x2.w;
    r[12] = (_Float16)x3.x; r[13] = (_Float16)x3.y;
    r[14] = (_Float16)x3.z; r[15] = (_Float16)x3.w;
    return r;
}

// ---------------------------------------------------------------------------
// WMMA GEMM: C[M x Nout] = A[M x K] * W^T, W row-major [Nout x K].
// One wave32 per (16*MT x 16) strip: B fragment loaded once per K-step,
// reused across MT M-tiles.  K is a template param -> fully unrolled.
// Requires: K % 32 == 0, Nout % 16 == 0, M % (16*MT) == 0, lda/ldw % 4 == 0.
// ---------------------------------------------------------------------------
template <int K, int MT>
__global__ __launch_bounds__(32)
void wmma_gemm_kernel(const float* __restrict__ A, const float* __restrict__ W,
                      float* __restrict__ C, int lda, int ldw, int ldc) {
    const int  lane = threadIdx.x;          // 0..31
    const int  lidx = lane & 15;
    const int  kb   = (lane >> 4) * 8;      // half-wave K offset (ISA layout)
    const long m0   = (long)blockIdx.y * (16 * MT);
    const int  col  = blockIdx.x * 16 + lidx;

    v8f acc[MT];
#pragma unroll
    for (int i = 0; i < MT; ++i) acc[i] = (v8f){};

    const float* wp = W + (long)col * ldw + kb;
    const float* ap = A + (m0 + lidx) * (long)lda + kb;

#pragma unroll
    for (int k0 = 0; k0 < K; k0 += 32) {
        v16h bf = load_frag(wp + k0);
        v16h af[MT];
#pragma unroll
        for (int i = 0; i < MT; ++i)
            af[i] = load_frag(ap + (long)i * 16 * lda + k0);
#pragma unroll
        for (int i = 0; i < MT; ++i)
            acc[i] = __builtin_amdgcn_wmma_f32_16x16x32_f16(
                false, af[i], false, bf, (short)0, acc[i], false, false);
    }

    // C/D layout: VGPR r holds row (tile_m + r + 8*(lane>>4)), lane&15 = col
    const int rbase = (lane >> 4) * 8;
#pragma unroll
    for (int i = 0; i < MT; ++i) {
        const long rm = m0 + i * 16 + rbase;
#pragma unroll
        for (int r = 0; r < 8; ++r)
            C[(rm + r) * (long)ldc + col] = acc[i][r];
    }
}

// ---------------------------------------------------------------------------
// Weight padding (removes all guards from the ragged GEMMs).
// ---------------------------------------------------------------------------
__global__ void pad_xproj_kernel(const float* __restrict__ in, float* __restrict__ out) {
    int i = blockIdx.x * blockDim.x + threadIdx.x;      // 48*384
    if (i >= XDBL_LD * D_INNER) return;
    int r = i / D_INNER, c = i % D_INNER;
    out[i] = (r < DT_RANK + 2 * D_STATE) ? in[r * D_INNER + c] : 0.0f;
}

__global__ void pad_dtw_kernel(const float* __restrict__ in, float* __restrict__ out) {
    int i = blockIdx.x * blockDim.x + threadIdx.x;      // 384*32
    if (i >= D_INNER * 32) return;
    int r = i / 32, c = i % 32;
    out[i] = (c < DT_RANK) ? in[r * DT_RANK + c] : 0.0f;
}

// ---------------------------------------------------------------------------
// Residual accumulate + LayerNorm.  One wave per token (192 = 6 vals/lane).
// ---------------------------------------------------------------------------
__global__ __launch_bounds__(256)
void ln_kernel(const float* __restrict__ base, const float* __restrict__ add,
               float* __restrict__ resid_out,
               const float* __restrict__ w, const float* __restrict__ b,
               float* __restrict__ out, int ntok) {
    const int wave = blockIdx.x * (blockDim.x >> 5) + (threadIdx.x >> 5);
    const int lane = threadIdx.x & 31;
    if (wave >= ntok) return;
    const long off = (long)wave * D_MODEL;

    float v[6];
    float s = 0.0f;
#pragma unroll
    for (int i = 0; i < 6; ++i) {
        float t = base[off + lane + 32 * i];
        if (add) t += add[off + lane + 32 * i];
        v[i] = t;
        s += t;
    }
#pragma unroll
    for (int o = 16; o > 0; o >>= 1) s += __shfl_xor(s, o, 32);
    const float mu = s / (float)D_MODEL;

    float var = 0.0f;
#pragma unroll
    for (int i = 0; i < 6; ++i) { float d = v[i] - mu; var += d * d; }
#pragma unroll
    for (int o = 16; o > 0; o >>= 1) var += __shfl_xor(var, o, 32);
    const float rstd = rsqrtf(var / (float)D_MODEL + 1e-5f);

#pragma unroll
    for (int i = 0; i < 6; ++i) {
        const int c = lane + 32 * i;
        if (resid_out) resid_out[off + c] = v[i];
        out[off + c] = (v[i] - mu) * rstd * w[c] + b[c];
    }
}

// ---------------------------------------------------------------------------
// Depthwise causal conv (width 4) + bias + SiLU.  One thread per (token, d).
// ---------------------------------------------------------------------------
__global__ __launch_bounds__(256)
void conv_silu_kernel(const float* __restrict__ xz, const float* __restrict__ cw,
                      const float* __restrict__ cb, float* __restrict__ xc) {
    const long idx = (long)blockIdx.x * blockDim.x + threadIdx.x;
    if (idx >= (long)NTOK * D_INNER) return;
    const int  d  = (int)(idx % D_INNER);
    const long bt = idx / D_INNER;
    const int  t  = (int)(bt % SEQLEN);
    const long bbase = bt - t;  // b*SEQLEN

    float acc = cb[d];
#pragma unroll
    for (int j = 0; j < D_CONV; ++j) {
        int tt = t - (D_CONV - 1) + j;
        if (tt >= 0)
            acc += cw[d * D_CONV + j] * xz[(bbase + tt) * (2L * D_INNER) + d];
    }
    xc[bt * D_INNER + d] = acc / (1.0f + __expf(-acc));
}

// ---------------------------------------------------------------------------
// dt = softplus(dt_raw + dt_bias[d])  (in place)
// ---------------------------------------------------------------------------
__global__ __launch_bounds__(256)
void softplus_kernel(float* __restrict__ dt, const float* __restrict__ bias) {
    const long idx = (long)blockIdx.x * blockDim.x + threadIdx.x;
    if (idx >= (long)NTOK * D_INNER) return;
    const int d = (int)(idx % D_INNER);
    float x = dt[idx] + bias[d];
    dt[idx] = (x > 20.0f) ? x : log1pf(__expf(x));
}

// ---------------------------------------------------------------------------
// Selective scan: one thread per (b, d), 16 states in registers, 2048 steps.
// Fused epilogue: y = (scan + u*D) * silu(z).
// ---------------------------------------------------------------------------
__global__ __launch_bounds__(128)
void scan_kernel(const float* __restrict__ dt, const float* __restrict__ xdbl,
                 const float* __restrict__ xc, const float* __restrict__ xz,
                 const float* __restrict__ A_log, const float* __restrict__ Dp,
                 float* __restrict__ y) {
    const int d = blockIdx.x * blockDim.x + threadIdx.x;
    const int b = blockIdx.y;
    if (d >= D_INNER) return;

    float Arow[D_STATE];
#pragma unroll
    for (int s = 0; s < D_STATE; ++s) Arow[s] = -__expf(A_log[d * D_STATE + s]);
    const float Dv = Dp[d];

    float h[D_STATE];
#pragma unroll
    for (int s = 0; s < D_STATE; ++s) h[s] = 0.0f;

    const long tokBase = (long)b * SEQLEN;
    for (int t = 0; t < SEQLEN; ++t) {
        const long tok = tokBase + t;
        const float dtv = dt[tok * D_INNER + d];
        const float u   = xc[tok * D_INNER + d];
        const float du  = dtv * u;
        const float* bd = xdbl + tok * XDBL_LD;
        float ys = 0.0f;
#pragma unroll
        for (int s = 0; s < D_STATE; ++s) {
            const float Bv = bd[DT_RANK + s];
            const float Cv = bd[DT_RANK + D_STATE + s];
            h[s] = __expf(dtv * Arow[s]) * h[s] + du * Bv;
            ys += h[s] * Cv;
        }
        const float zv = xz[tok * (2L * D_INNER) + D_INNER + d];
        y[tok * D_INNER + d] = (ys + u * Dv) * (zv / (1.0f + __expf(-zv)));
    }
}

// ---------------------------------------------------------------------------
extern "C" void kernel_launch(void* const* d_in, const int* in_sizes, int n_in,
                              void* d_out, int out_size, void* d_ws, size_t ws_size,
                              hipStream_t stream) {
    (void)in_sizes; (void)n_in; (void)out_size; (void)ws_size;

    const float* x         = (const float*)d_in[0];
    const float* in_proj_w = (const float*)d_in[1];   // [4, 768, 192]
    const float* conv_w    = (const float*)d_in[2];   // [4, 384, 4]
    const float* conv_b    = (const float*)d_in[3];   // [4, 384]
    const float* x_proj_w  = (const float*)d_in[4];   // [4, 44, 384]
    const float* dt_proj_w = (const float*)d_in[5];   // [4, 384, 12]
    const float* dt_proj_b = (const float*)d_in[6];   // [4, 384]
    const float* A_log     = (const float*)d_in[7];   // [4, 384, 16]
    const float* D_param   = (const float*)d_in[8];   // [4, 384]
    const float* out_proj_w= (const float*)d_in[9];   // [4, 192, 384]
    const float* norm_w    = (const float*)d_in[10];  // [4, 192]
    const float* norm_b    = (const float*)d_in[11];  // [4, 192]
    const float* normf_w   = (const float*)d_in[12];  // [192]
    const float* normf_b   = (const float*)d_in[13];  // [192]
    float* out = (float*)d_out;

    const long N = NTOK;
    float* ws       = (float*)d_ws;
    float* residual = ws;                       // N*192
    float* hbuf     = residual + N * 192;       // N*192
    float* xz       = hbuf + N * 192;           // N*768
    float* xc       = xz + N * 768;             // N*384
    float* xdbl     = xc + N * 384;             // N*48 (padded)
    float* dtb      = xdbl + N * XDBL_LD;       // N*384
    float* yb       = dtb + N * 384;            // N*384
    float* hidden   = yb + N * 384;             // N*192
    float* wpad_x   = hidden + N * 192;         // 48*384
    float* wpad_dt  = wpad_x + XDBL_LD * D_INNER; // 384*32

    const int lnBlocks = NTOK / 8;                       // 8 waves/block
    const long elemInner = (long)NTOK * D_INNER;
    const int ewBlocks = (int)((elemInner + 255) / 256);
    const int mStrips  = NTOK / 64;                      // 16*MT rows per strip

    for (int layer = 0; layer < DEPTH; ++layer) {
        const float* ipw    = in_proj_w  + (long)layer * (2 * D_INNER) * D_MODEL;
        const float* cw     = conv_w     + (long)layer * D_INNER * D_CONV;
        const float* cb     = conv_b     + (long)layer * D_INNER;
        const float* xpw    = x_proj_w   + (long)layer * (DT_RANK + 2 * D_STATE) * D_INNER;
        const float* dtw    = dt_proj_w  + (long)layer * D_INNER * DT_RANK;
        const float* dtbias = dt_proj_b  + (long)layer * D_INNER;
        const float* al     = A_log      + (long)layer * D_INNER * D_STATE;
        const float* dp     = D_param    + (long)layer * D_INNER;
        const float* opw    = out_proj_w + (long)layer * D_MODEL * D_INNER;
        const float* nw     = norm_w     + (long)layer * D_MODEL;
        const float* nb     = norm_b     + (long)layer * D_MODEL;

        // zero-padded copies of the ragged weights
        pad_xproj_kernel<<<(XDBL_LD * D_INNER + 255) / 256, 256, 0, stream>>>(xpw, wpad_x);
        pad_dtw_kernel<<<(D_INNER * 32 + 255) / 256, 256, 0, stream>>>(dtw, wpad_dt);

        // residual (+= hidden) ; h = LN(residual)
        ln_kernel<<<lnBlocks, 256, 0, stream>>>(
            layer == 0 ? x : residual, layer == 0 ? nullptr : hidden,
            residual, nw, nb, hbuf, NTOK);

        // xz = h @ in_proj_w^T   [N,192]x[768,192] -> [N,768]
        wmma_gemm_kernel<D_MODEL, 4><<<dim3(48, mStrips), 32, 0, stream>>>(
            hbuf, ipw, xz, D_MODEL, D_MODEL, 2 * D_INNER);

        // xc = silu(conv(xp) + bias)
        conv_silu_kernel<<<ewBlocks, 256, 0, stream>>>(xz, cw, cb, xc);

        // x_dbl = xc @ x_proj_w_pad^T   [N,384]x[48,384] -> [N,48]
        wmma_gemm_kernel<D_INNER, 4><<<dim3(3, mStrips), 32, 0, stream>>>(
            xc, wpad_x, xdbl, D_INNER, D_INNER, XDBL_LD);

        // dt_raw = x_dbl[:, :32] @ dt_proj_w_pad^T   [N,32]x[384,32] -> [N,384]
        wmma_gemm_kernel<32, 4><<<dim3(24, mStrips), 32, 0, stream>>>(
            xdbl, wpad_dt, dtb, XDBL_LD, 32, D_INNER);

        // dt = softplus(dt_raw + bias)
        softplus_kernel<<<ewBlocks, 256, 0, stream>>>(dtb, dtbias);

        // selective scan (sequential over L), fused D-skip and silu(z) gate
        scan_kernel<<<dim3(3, BATCH), 128, 0, stream>>>(
            dtb, xdbl, xc, xz, al, dp, yb);

        // hidden = y @ out_proj_w^T   [N,384]x[192,384] -> [N,192]
        wmma_gemm_kernel<D_INNER, 4><<<dim3(12, mStrips), 32, 0, stream>>>(
            yb, opw, hidden, D_INNER, D_INNER, D_MODEL);
    }

    // final: residual += hidden ; out = LN(residual, normf)
    ln_kernel<<<lnBlocks, 256, 0, stream>>>(
        residual, hidden, nullptr, normf_w, normf_b, out, NTOK);
}
